// TrueHigherOrderAttention_62551903699153
// MI455X (gfx1250) — matrix operand order repaired and checked by reading, compiled, not run on hardware
//
#include <hip/hip_runtime.h>

// ---------------------------------------------------------------------------
// Order-3 attention for MI455X (gfx1250, wave32).
// WMMA f32_16x16x32_f16 for all GEMMs; K/V panels staged to LDS with
// GLOBAL_LOAD_ASYNC_TO_LDS_B128 (ASYNCcnt) in the T^3 passes.
// B=2, T=192, C=512, nh=8, hd=64.
// ---------------------------------------------------------------------------

typedef __attribute__((ext_vector_type(16))) _Float16 v16h;
typedef __attribute__((ext_vector_type(8)))  _Float16 h8;
typedef __attribute__((ext_vector_type(8)))  float    v8f;

#define DEVINL __device__ __forceinline__

#define LDS_LD 72   // padded row stride (halves): 144B -> 36-dword bank stride

DEVINL int lane_id() { return (int)(threadIdx.x & 31); }

// ---- WMMA wrapper: D = A(16x32 f16) x B(32x16 f16) + C(16x16 f32) ----------
DEVINL v8f wmma_f16(v16h a, v16h b, v8f c) {
  return __builtin_amdgcn_wmma_f32_16x16x32_f16(
      /*neg_a=*/false, a, /*neg_b=*/false, b,
      /*c_mod=*/(short)0, c, /*reuse_a=*/false, /*reuse_b=*/false);
}

// ---- CDNA5 async global->LDS copy (per-lane, 16B), tracked by ASYNCcnt -----
DEVINL void async_g2l_b128(unsigned lds_addr, const void* gaddr) {
  asm volatile("global_load_async_to_lds_b128 %0, %1, off"
               :: "v"(lds_addr), "v"(gaddr)
               : "memory");
}
DEVINL void wait_async_zero() {
  asm volatile("s_wait_asynccnt 0" ::: "memory");
}
DEVINL unsigned lds_off(const void* p) {   // LDS generic addr: offset in [31:0]
  return (unsigned)(unsigned long long)p;
}

// ---- A-fragment loader (16-bit A 16x32 ISA layout) -------------------------
// Lane holds row `row`; K chunks: lanes 0-15 -> {c0+0..7, c0+16..23},
// lanes 16-31 -> {c0+8..15, c0+24..31}. v16h elem e -> VGPR e/2 half e%2.
DEVINL v16h load_fragA(const _Float16* __restrict__ base, int ld, int row, int c0) {
  const _Float16* p = base + (long)row * ld + c0 + ((lane_id() >> 4) << 3);
  h8 lo = *(const h8*)(p);
  h8 hi = *(const h8*)(p + 16);
  v16h r;
#pragma unroll
  for (int e = 0; e < 8; ++e) { r[e] = lo[e]; r[e + 8] = hi[e]; }
  return r;
}

// ---- B-fragment loader (16-bit B 32x16 ISA layout) -------------------------
// Lane holds column n (= source row, since B[k,n] = Src[n,k]);
// K: lanes 0-15 -> {c0+0..7, c0+16..23}, lanes 16-31 -> {c0+8..15, c0+24..31}.
DEVINL v16h load_fragB(const _Float16* __restrict__ base, int ld, int row, int c0) {
  const _Float16* p = base + (long)row * ld + c0 + ((lane_id() >> 4) << 3);
  h8 lo = *(const h8*)(p);
  h8 hi = *(const h8*)(p + 16);
  v16h r;
#pragma unroll
  for (int e = 0; e < 8; ++e) { r[e] = lo[e]; r[e + 8] = hi[e]; }
  return r;
}

// ---------------------------------------------------------------------------
// Kernel 0: f32 -> f16 conversion
// ---------------------------------------------------------------------------
__global__ void cvt_f32_f16(const float* __restrict__ s, _Float16* __restrict__ d, int n) {
  int i = blockIdx.x * blockDim.x + threadIdx.x;
  if (i < n) d[i] = (_Float16)s[i];
}

// ---------------------------------------------------------------------------
// Kernel 1: projections.  out[m,n] = sum_c x[m,c] * W[n,c]  (m=b*192+t, n=h*64+d)
// Stored to (bh, t, d) layout.  proj 0,1,2 -> Qh,Kh,Vh (Vh also f32 copy).
// ---------------------------------------------------------------------------
__global__ void proj_gemm(const _Float16* __restrict__ xh,
                          const _Float16* __restrict__ W0h,
                          const _Float16* __restrict__ W1h,
                          const _Float16* __restrict__ W2h,
                          _Float16* __restrict__ Qh,
                          _Float16* __restrict__ Kh,
                          _Float16* __restrict__ Vh,
                          float* __restrict__ Vf) {
  int wv = threadIdx.x >> 5;
  int tile = blockIdx.x * (blockDim.x >> 5) + wv;
  if (tile >= 3 * 24 * 32) return;           // wave-uniform
  int proj = tile / (24 * 32);
  int rem  = tile % (24 * 32);
  int tm = rem / 32, tn = rem % 32;
  const _Float16* Wh = (proj == 0) ? W0h : ((proj == 1) ? W1h : W2h);
  int lane = lane_id();
  int m0 = tm * 16, n0 = tn * 16;
  v8f acc = {};
#pragma unroll 4
  for (int c0 = 0; c0 < 512; c0 += 32) {
    v16h a = load_fragA(xh, 512, m0 + (lane & 15), c0);
    v16h b = load_fragB(Wh, 512, n0 + (lane & 15), c0);
    acc = wmma_f16(a, b, acc);
  }
  _Float16* Out = (proj == 0) ? Qh : ((proj == 1) ? Kh : Vh);
#pragma unroll
  for (int r = 0; r < 8; ++r) {
    int M = m0 + r + ((lane >> 4) << 3);     // global token row
    int N = n0 + (lane & 15);                // channel
    int b_ = M / 192, t = M % 192;
    int h  = N >> 6,  d = N & 63;
    long idx = (((long)(b_ * 8 + h)) * 192 + t) * 64 + d;
    float v = acc[r];
    Out[idx] = (_Float16)v;
    if (proj == 2) Vf[idx] = v;
  }
}

// ---------------------------------------------------------------------------
// Stage the (b,h) K and V panels (192x64 f16) to padded LDS with async copies.
// 1536 16-byte chunks per panel, 512 threads -> 3 chunks each per panel.
// ---------------------------------------------------------------------------
DEVINL void stage_panels(const _Float16* __restrict__ Kg,
                         const _Float16* __restrict__ Vg,
                         _Float16 (*Ks)[LDS_LD], _Float16 (*Vs)[LDS_LD]) {
  int tid = (int)threadIdx.x;           // 512 threads
#pragma unroll
  for (int c = 0; c < 3; ++c) {
    int chunk = tid + c * 512;          // 0..1535
    int row = chunk >> 3;
    int col = (chunk & 7) << 3;         // half offset within row
    async_g2l_b128(lds_off(&Ks[row][col]), Kg + row * 64 + col);
    async_g2l_b128(lds_off(&Vs[row][col]), Vg + row * 64 + col);
  }
  wait_async_zero();                    // this wave's asyncs done
  __syncthreads();                      // all waves' asyncs done
}

#define SCALE 0.125f   // 1/sqrt(64)

// ---------------------------------------------------------------------------
// Kernel 2: pass 1 — per-(bh,i) online max / sum-exp over masked scores.
// Grid: 192 blocks (bh * 12 i-tiles), 512 threads = 16 waves, wave wv -> i.
// s[i,j,k] = sum_d (q_i[d]*k_j[d]) * v_k[d]  ==  (q_i (.) K) @ V^T
// ---------------------------------------------------------------------------
__global__ void hoa_pass1(const _Float16* __restrict__ Qh,
                          const _Float16* __restrict__ Kh,
                          const _Float16* __restrict__ Vh,
                          float* __restrict__ mz) {
  __shared__ _Float16 Ks[192][LDS_LD];
  __shared__ _Float16 Vs[192][LDS_LD];
  int bh = blockIdx.x / 12;
  int ti = blockIdx.x % 12;
  int wv = threadIdx.x >> 5;
  int i  = ti * 16 + wv;
  int lane = lane_id();
  const _Float16* Q = Qh + (long)bh * 192 * 64;

  stage_panels(Kh + (long)bh * 192 * 64, Vh + (long)bh * 192 * 64, Ks, Vs);

  v16h q0 = load_fragA(Q, 64, i, 0);
  v16h q1 = load_fragA(Q, 64, i, 32);

  float m = -__builtin_inff(), z = 0.0f;
  int tjmax = i >> 4;
  for (int tj = 0; tj <= tjmax; ++tj) {
    v16h a0 = load_fragA(&Ks[0][0], LDS_LD, tj * 16 + (lane & 15), 0)  * q0;
    v16h a1 = load_fragA(&Ks[0][0], LDS_LD, tj * 16 + (lane & 15), 32) * q1;
    for (int tk = 0; tk <= tj; ++tk) {
      v16h b0 = load_fragB(&Vs[0][0], LDS_LD, tk * 16 + (lane & 15), 0);
      v16h b1 = load_fragB(&Vs[0][0], LDS_LD, tk * 16 + (lane & 15), 32);
      v8f c = {};
      c = wmma_f16(a0, b0, c);
      c = wmma_f16(a1, b1, c);
#pragma unroll
      for (int r = 0; r < 8; ++r) {
        int j = tj * 16 + r + ((lane >> 4) << 3);
        int k = tk * 16 + (lane & 15);
        if (j <= i && k <= j) {
          float s = c[r] * SCALE;
          if (s > m) { z *= __expf(m - s); m = s; }
          z += __expf(s - m);
        }
      }
    }
  }
  // wave32 (m,z) reduction
#pragma unroll
  for (int off = 16; off; off >>= 1) {
    float m2 = __shfl_xor(m, off);
    float z2 = __shfl_xor(z, off);
    float M  = fmaxf(m, m2);
    float e1 = (m  >= M) ? 1.0f : __expf(m  - M);
    float e2 = (m2 >= M) ? 1.0f : __expf(m2 - M);
    z = z * e1 + z2 * e2;
    m = M;
  }
  if (lane == 0) {
    mz[(bh * 192 + i) * 2 + 0] = m;
    mz[(bh * 192 + i) * 2 + 1] = z;
  }
}

// ---------------------------------------------------------------------------
// Kernel 3: combine per-i (m,z) into global (M,Z) per (b,h) — whole-T^3 softmax.
// ---------------------------------------------------------------------------
__global__ void hoa_reduce(const float* __restrict__ mz, float* __restrict__ MZ) {
  int bh = blockIdx.x;
  int lane = (int)threadIdx.x;         // 32 threads = 1 wave
  float m = -__builtin_inff(), z = 0.0f;
  for (int i = lane; i < 192; i += 32) {
    float mi = mz[(bh * 192 + i) * 2 + 0];
    float zi = mz[(bh * 192 + i) * 2 + 1];
    float M  = fmaxf(m, mi);
    float e1 = (m  >= M) ? 1.0f : __expf(m  - M);
    float e2 = (mi >= M) ? 1.0f : __expf(mi - M);
    z = z * e1 + zi * e2;
    m = M;
  }
#pragma unroll
  for (int off = 16; off; off >>= 1) {
    float m2 = __shfl_xor(m, off);
    float z2 = __shfl_xor(z, off);
    float M  = fmaxf(m, m2);
    float e1 = (m  >= M) ? 1.0f : __expf(m  - M);
    float e2 = (m2 >= M) ? 1.0f : __expf(m2 - M);
    z = z * e1 + z2 * e2;
    m = M;
  }
  if (lane == 0) { MZ[bh * 2 + 0] = m; MZ[bh * 2 + 1] = z; }
}

// ---------------------------------------------------------------------------
// Kernel 4: pass 2 — recompute scores, p=exp(s-M), accumulate column sums
// c_i[k] = sum_j p[i,j,k], then y_i = (c_i @ V) / Z  (V in f32 for accuracy).
// ---------------------------------------------------------------------------
__global__ void hoa_pass2(const _Float16* __restrict__ Qh,
                          const _Float16* __restrict__ Kh,
                          const _Float16* __restrict__ Vh,
                          const float* __restrict__ Vf,
                          const float* __restrict__ MZ,
                          _Float16* __restrict__ yh) {
  __shared__ _Float16 Ks[192][LDS_LD];
  __shared__ _Float16 Vs[192][LDS_LD];
  __shared__ float cs[16][192];
  int bh = blockIdx.x / 12;
  int ti = blockIdx.x % 12;
  int wv = threadIdx.x >> 5;
  int i  = ti * 16 + wv;
  int lane = lane_id();
  const _Float16* Q = Qh + (long)bh * 192 * 64;
  float Mg   = MZ[bh * 2 + 0];
  float Zinv = 1.0f / MZ[bh * 2 + 1];

  stage_panels(Kh + (long)bh * 192 * 64, Vh + (long)bh * 192 * 64, Ks, Vs);

  v16h q0 = load_fragA(Q, 64, i, 0);
  v16h q1 = load_fragA(Q, 64, i, 32);

  float cacc[12];
#pragma unroll
  for (int t = 0; t < 12; ++t) cacc[t] = 0.0f;

  int tjmax = i >> 4;
  for (int tj = 0; tj <= tjmax; ++tj) {
    v16h a0 = load_fragA(&Ks[0][0], LDS_LD, tj * 16 + (lane & 15), 0)  * q0;
    v16h a1 = load_fragA(&Ks[0][0], LDS_LD, tj * 16 + (lane & 15), 32) * q1;
    for (int tk = 0; tk <= tj; ++tk) {
      v16h b0 = load_fragB(&Vs[0][0], LDS_LD, tk * 16 + (lane & 15), 0);
      v16h b1 = load_fragB(&Vs[0][0], LDS_LD, tk * 16 + (lane & 15), 32);
      v8f c = {};
      c = wmma_f16(a0, b0, c);
      c = wmma_f16(a1, b1, c);
      float col = 0.0f;
#pragma unroll
      for (int r = 0; r < 8; ++r) {
        int j = tj * 16 + r + ((lane >> 4) << 3);
        int k = tk * 16 + (lane & 15);
        if (j <= i && k <= j) col += __expf(c[r] * SCALE - Mg);
      }
      col += __shfl_xor(col, 16);   // fold j halves; lanes L, L^16 now equal
      cacc[tk] += col;
    }
  }
#pragma unroll
  for (int tk = 0; tk < 12; ++tk) cs[wv][tk * 16 + (lane & 15)] = cacc[tk];
  __syncthreads();

  // y_i[d] = (sum_k c[k] * Vf[k,d]) / Z ; each lane owns d and d+32
  int d0 = lane, d1 = lane + 32;
  const float* Vr = Vf + (long)bh * 192 * 64;
  float y0 = 0.0f, y1 = 0.0f;
  for (int k = 0; k <= i; ++k) {
    float ck = cs[wv][k];
    y0 = fmaf(ck, Vr[k * 64 + d0], y0);
    y1 = fmaf(ck, Vr[k * 64 + d1], y1);
  }
  y0 *= Zinv; y1 *= Zinv;
  int b_ = bh >> 3, h = bh & 7;
  long orow = ((long)(b_ * 192 + i)) * 512 + h * 64;  // (B,T,C) layout
  yh[orow + d0] = (_Float16)y0;
  yh[orow + d1] = (_Float16)y1;
}

// ---------------------------------------------------------------------------
// Kernel 5: out = y @ Wp^T, f32 output.
// ---------------------------------------------------------------------------
__global__ void out_gemm(const _Float16* __restrict__ yh,
                         const _Float16* __restrict__ Wph,
                         float* __restrict__ out) {
  int wv = threadIdx.x >> 5;
  int tile = blockIdx.x * (blockDim.x >> 5) + wv;
  if (tile >= 24 * 32) return;               // wave-uniform
  int tm = tile / 32, tn = tile % 32;
  int lane = lane_id();
  v8f acc = {};
#pragma unroll 4
  for (int c0 = 0; c0 < 512; c0 += 32) {
    v16h a = load_fragA(yh,  512, tm * 16 + (lane & 15), c0);
    v16h b = load_fragB(Wph, 512, tn * 16 + (lane & 15), c0);
    acc = wmma_f16(a, b, acc);
  }
#pragma unroll
  for (int r = 0; r < 8; ++r) {
    int M = tm * 16 + r + ((lane >> 4) << 3);
    int N = tn * 16 + (lane & 15);
    out[(long)M * 512 + N] = acc[r];
  }
}

// ---------------------------------------------------------------------------
extern "C" void kernel_launch(void* const* d_in, const int* in_sizes, int n_in,
                              void* d_out, int out_size, void* d_ws, size_t ws_size,
                              hipStream_t stream) {
  const float* x  = (const float*)d_in[0];
  const float* W0 = (const float*)d_in[1];
  const float* W1 = (const float*)d_in[2];
  const float* W2 = (const float*)d_in[3];
  const float* Wp = (const float*)d_in[4];

  const int NX = 2 * 192 * 512;   // 196608
  const int NW = 512 * 512;       // 262144

  char* ws = (char*)d_ws;
  size_t off = 0;
  auto alloc = [&](size_t bytes) -> void* {
    void* p = ws + off;
    off += (bytes + 255) & ~(size_t)255;
    return p;
  };
  _Float16* xh  = (_Float16*)alloc((size_t)NX * 2);
  _Float16* W0h = (_Float16*)alloc((size_t)NW * 2);
  _Float16* W1h = (_Float16*)alloc((size_t)NW * 2);
  _Float16* W2h = (_Float16*)alloc((size_t)NW * 2);
  _Float16* Wph = (_Float16*)alloc((size_t)NW * 2);
  _Float16* Qh  = (_Float16*)alloc((size_t)NX * 2);
  _Float16* Kh  = (_Float16*)alloc((size_t)NX * 2);
  _Float16* Vh  = (_Float16*)alloc((size_t)NX * 2);
  float*    Vf  = (float*)   alloc((size_t)NX * 4);
  _Float16* yh  = (_Float16*)alloc((size_t)NX * 2);
  float*    mz  = (float*)   alloc((size_t)16 * 192 * 2 * 4);
  float*    MZ  = (float*)   alloc((size_t)16 * 2 * 4);
  (void)ws_size; (void)in_sizes; (void)n_in; (void)out_size;

  cvt_f32_f16<<<(NX + 255) / 256, 256, 0, stream>>>(x,  xh,  NX);
  cvt_f32_f16<<<(NW + 255) / 256, 256, 0, stream>>>(W0, W0h, NW);
  cvt_f32_f16<<<(NW + 255) / 256, 256, 0, stream>>>(W1, W1h, NW);
  cvt_f32_f16<<<(NW + 255) / 256, 256, 0, stream>>>(W2, W2h, NW);
  cvt_f32_f16<<<(NW + 255) / 256, 256, 0, stream>>>(Wp, Wph, NW);

  // 2304 tiles / 8 waves per 256-thread block
  proj_gemm<<<288, 256, 0, stream>>>(xh, W0h, W1h, W2h, Qh, Kh, Vh, Vf);

  hoa_pass1<<<192, 512, 0, stream>>>(Qh, Kh, Vh, mz);
  hoa_reduce<<<16, 32, 0, stream>>>(mz, MZ);
  hoa_pass2<<<192, 512, 0, stream>>>(Qh, Kh, Vh, Vf, MZ, yh);

  // 768 tiles / 8 waves per block
  out_gemm<<<96, 256, 0, stream>>>(yh, Wph, (float*)d_out);
}
